// SO2_Linear_49529562858341
// MI455X (gfx1250) — matrix operand order repaired and checked by reading, not compiled, not run on hardware
//
#include <hip/hip_runtime.h>
#include <hip/hip_bf16.h>
#include <math.h>

typedef _Float16 half16 __attribute__((ext_vector_type(16)));
typedef _Float16 half8  __attribute__((ext_vector_type(8)));
typedef float    f32x8  __attribute__((ext_vector_type(8)));

#define L_MAX     4
#define D_TOT     25          // (L_MAX+1)^2
#define C_IN      128
#define C_OUT     128
#define N_EDGES   16384
#define N_GRAPHS  8
#define N_EXPERTS 4
#define EDGES_PER_G (N_EDGES / N_GRAPHS)   // 2048
#define DW_STRIDE 165                      // sum of (2l+1)^2

// per-l base offset into the G/H constant table (floats)
__device__ __constant__ int c_ghBase[5] = {0, 2, 38, 188, 580};
#define GH_TOTAL 1390

// ---------------------------------------------------------------------------
// Kernel 1: build Ry spectral constants G_m, H_m per l on device (double math)
//   Ry_l(beta) = sum_{m=0..l} cos(m*beta)*G_m + sin(m*beta)*H_m
//   G_m = 2*Re(P_m) (m>0) / Re(P_0),  H_m = -2*Im(P_m)  (H_0 ~ 0)
//   P_m = prod_{m'!=m} (A - i m' I) / (i (m - m')),  A real generator.
// ---------------------------------------------------------------------------
__global__ void init_ry_consts(float* gh) {
    int task = threadIdx.x;
    if (task >= 15) return;                 // (l,m): 1+2+3+4+5 = 15 tasks
    int l = 0, t = task;
    while (t > l) { t -= (l + 1); ++l; }
    int m = t;
    const int d = 2 * l + 1;

    double jp[9][9];
    for (int i = 0; i < d; ++i) for (int j = 0; j < d; ++j) jp[i][j] = 0.0;
    for (int i = 0; i < d - 1; ++i) {
        double mi = (double)(-l + i);
        jp[i + 1][i] = sqrt((double)l * (l + 1) - mi * (mi + 1.0));
    }
    // M = -i*jy = (jp^T - jp)/2  (real)
    double M[9][9];
    for (int i = 0; i < d; ++i)
        for (int j = 0; j < d; ++j)
            M[i][j] = 0.5 * (jp[j][i] - jp[i][j]);
    // B (complex -> real SH change of basis)
    double Br[9][9], Bi[9][9];
    for (int i = 0; i < d; ++i) for (int j = 0; j < d; ++j) { Br[i][j] = 0.0; Bi[i][j] = 0.0; }
    const double is2 = 1.0 / sqrt(2.0);
    for (int k = 0; k < d; ++k) {
        int mk = l - k;
        if (mk == 0) { Br[l][k] = 1.0; }
        else if (mk > 0) {
            Br[l - mk][k] = is2;
            Br[l + mk][k] = ((mk & 1) ? -1.0 : 1.0) * is2;
        } else {
            int mu = -mk;
            Bi[l - mu][k] = -is2;
            Bi[l + mu][k] = ((mu & 1) ? -1.0 : 1.0) * is2;
        }
    }
    // A = Re( B^dag M B )   (A is real)
    double Tr[9][9], Ti[9][9], A[9][9];
    for (int i = 0; i < d; ++i)
        for (int j = 0; j < d; ++j) {
            double sr = 0.0, si = 0.0;
            for (int k = 0; k < d; ++k) { sr += M[i][k] * Br[k][j]; si += M[i][k] * Bi[k][j]; }
            Tr[i][j] = sr; Ti[i][j] = si;
        }
    for (int i = 0; i < d; ++i)
        for (int j = 0; j < d; ++j) {
            double s = 0.0;
            for (int k = 0; k < d; ++k) s += Br[k][i] * Tr[k][j] + Bi[k][i] * Ti[k][j];
            A[i][j] = s;
        }
    // P_m via polynomial product over known eigenvalues i*m'
    double Pr[9][9], Pi[9][9], Qr[9][9], Qi[9][9];
    for (int i = 0; i < d; ++i)
        for (int j = 0; j < d; ++j) { Pr[i][j] = (i == j) ? 1.0 : 0.0; Pi[i][j] = 0.0; }
    for (int mp = -l; mp <= l; ++mp) {
        if (mp == m) continue;
        for (int i = 0; i < d; ++i)
            for (int j = 0; j < d; ++j) {
                double ar = 0.0, ai = 0.0;
                for (int k = 0; k < d; ++k) { ar += A[i][k] * Pr[k][j]; ai += A[i][k] * Pi[k][j]; }
                // (A - i*mp I) P :  -i*mp*(Pr + i Pi) = mp*Pi - i*mp*Pr
                ar += (double)mp * Pi[i][j];
                ai -= (double)mp * Pr[i][j];
                Qr[i][j] = ar; Qi[i][j] = ai;
            }
        double inv = 1.0 / (double)(m - mp);
        // multiply by 1/(i*(m-mp)) = -i*inv
        for (int i = 0; i < d; ++i)
            for (int j = 0; j < d; ++j) {
                Pr[i][j] = Qi[i][j] * inv;
                Pi[i][j] = -Qr[i][j] * inv;
            }
    }
    float scale = (m == 0) ? 1.0f : 2.0f;
    float* G = gh + c_ghBase[l] + (2 * m) * d * d;
    float* H = G + d * d;
    for (int i = 0; i < d; ++i)
        for (int j = 0; j < d; ++j) {
            G[i * d + j] = scale * (float)Pr[i][j];
            H[i * d + j] = -scale * (float)Pi[i][j];
        }
}

// ---------------------------------------------------------------------------
// Kernel 2: per-edge Wigner blocks  D_l = Rz(alpha) Ry(beta) Rz(gamma)
// ---------------------------------------------------------------------------
__global__ void wigner_kernel(const float* __restrict__ alpha,
                              const float* __restrict__ beta,
                              const float* __restrict__ gamma,
                              const float* __restrict__ gh,
                              float* __restrict__ Dw) {
    int e = blockIdx.x * blockDim.x + threadIdx.x;
    if (e >= N_EDGES) return;
    float a = alpha[e], b = beta[e], g = gamma[e];
    float* out = Dw + (size_t)e * DW_STRIDE;
    int dOff = 0;
    for (int l = 0; l <= L_MAX; ++l) {
        int d = 2 * l + 1;
        float Ry[81], T[81];
        const float* ghl = gh + c_ghBase[l];
        for (int i = 0; i < d * d; ++i) Ry[i] = 0.f;
        for (int m = 0; m <= l; ++m) {
            float cm = cosf((float)m * b), sm = sinf((float)m * b);
            const float* G = ghl + (2 * m) * d * d;
            const float* H = G + d * d;
            for (int i = 0; i < d * d; ++i) Ry[i] += cm * G[i] + sm * H[i];
        }
        // T = Rz(alpha) * Ry   (Rz sparse: diag cos(freq*a), antidiag sin(freq*a))
        for (int i = 0; i < d; ++i) {
            float ca = cosf((float)(l - i) * a), sa = sinf((float)(l - i) * a);
            for (int j = 0; j < d; ++j)
                T[i * d + j] = ca * Ry[i * d + j] + sa * Ry[(d - 1 - i) * d + j];
        }
        // D = T * Rz(gamma)
        for (int j = 0; j < d; ++j) {
            float cg = cosf((float)(l - j) * g);
            float sg = sinf((float)(j - l) * g);   // freq of row (d-1-j)
            for (int i = 0; i < d; ++i)
                out[dOff + i * d + j] = T[i * d + j] * cg + T[i * d + (d - 1 - j)] * sg;
        }
        dOff += d * d;
    }
}

// ---------------------------------------------------------------------------
// Kernel 3: rotate x with D_l and scatter into m-grouped f16 GEMM A-panels
//   X0:  (E, 640)             m=0 channels (l ordered)
//   Xm:  (E, 2, numl*128)     t=0 <- m'=-m rows, t=1 <- m'=+m rows
// block = 256 threads = 2 edges x 128 channels
// ---------------------------------------------------------------------------
__global__ void rot_gather_kernel(const float* __restrict__ x,
                                  const float* __restrict__ Dw,
                                  _Float16* __restrict__ X0,
                                  _Float16* __restrict__ X1,
                                  _Float16* __restrict__ X2,
                                  _Float16* __restrict__ X3,
                                  _Float16* __restrict__ X4) {
    __shared__ float sD[2 * DW_STRIDE];
    int e0 = blockIdx.x * 2;
    for (int i = threadIdx.x; i < 2 * DW_STRIDE; i += blockDim.x)
        sD[i] = Dw[(size_t)e0 * DW_STRIDE + i];
    __syncthreads();
    int le = threadIdx.x >> 7;
    int c = threadIdx.x & 127;
    int e = e0 + le;
    const float* xe = x + (size_t)e * D_TOT * C_IN;
    const float* De = sD + le * DW_STRIDE;
    _Float16* XmArr[5] = {X0, X1, X2, X3, X4};
    int dOff = 0;
    for (int l = 0; l <= L_MAX; ++l) {
        int d = 2 * l + 1;
        float xv[9];
        for (int j = 0; j < d; ++j) xv[j] = xe[(l * l + j) * C_IN + c];
        for (int i = 0; i < d; ++i) {
            float s = 0.f;
            for (int j = 0; j < d; ++j) s += De[dOff + i * d + j] * xv[j];
            int mm = i - l;
            if (mm == 0) {
                X0[(size_t)e * ((L_MAX + 1) * C_IN) + l * C_IN + c] = (_Float16)s;
            } else {
                int m = mm < 0 ? -mm : mm;
                int t = mm < 0 ? 0 : 1;
                int H = (L_MAX + 1 - m) * C_IN;
                XmArr[m][((size_t)e * 2 + t) * H + (l - m) * C_IN + c] = (_Float16)s;
            }
        }
        dOff += d * d;
    }
}

// ---------------------------------------------------------------------------
// Weight / bias mixing:  W_g = sum_k coeff[g,k]*W_k + W_shared  (f16 out)
// ---------------------------------------------------------------------------
__global__ void combine_w_f16(const float* __restrict__ coeff,
                              const float* __restrict__ experts,
                              const float* __restrict__ shared_,
                              _Float16* __restrict__ out, int sz) {
    int idx = blockIdx.x * blockDim.x + threadIdx.x;
    if (idx >= N_GRAPHS * sz) return;
    int g = idx / sz, p = idx - g * sz;
    float v = shared_[p];
    for (int k = 0; k < N_EXPERTS; ++k)
        v += coeff[g * N_EXPERTS + k] * experts[(size_t)k * sz + p];
    out[idx] = (_Float16)v;
}

__global__ void combine_b_f32(const float* __restrict__ coeff,
                              const float* __restrict__ experts,
                              const float* __restrict__ shared_,
                              float* __restrict__ out, int sz) {
    int idx = blockIdx.x * blockDim.x + threadIdx.x;
    if (idx >= N_GRAPHS * sz) return;
    int g = idx / sz, p = idx - g * sz;
    float v = shared_[p];
    for (int k = 0; k < N_EXPERTS; ++k)
        v += coeff[g * N_EXPERTS + k] * experts[(size_t)k * sz + p];
    out[idx] = v;
}

// ---------------------------------------------------------------------------
// WMMA fragment helpers (CDNA5 v_wmma_f32_16x16x32_f16 VGPR layouts)
// A 16x32: lane<16 row=lane, halves = K[k0..k0+8) ++ K[k0+16..k0+24)
//          lane>=16 same row, halves shifted by +8
// B 32x16: lane holds column (lane&15); same K split as A:
//          K[k0..k0+8) ++ K[k0+16..k0+24), shifted by +8 for lane>=16
// ---------------------------------------------------------------------------
__device__ inline half16 cat8(half8 lo, half8 hi) {
    half16 r;
#pragma unroll
    for (int i = 0; i < 8; ++i) { r[i] = lo[i]; r[8 + i] = hi[i]; }
    return r;
}
__device__ inline half16 loadA(const _Float16* p) {   // p = rowbase + half*8 + k0
    return cat8(*(const half8*)p, *(const half8*)(p + 16));
}
__device__ inline half16 loadB(const _Float16* p) {   // p = rowbase + half*8 + k0
    return cat8(*(const half8*)p, *(const half8*)(p + 16));
}

// ---------------------------------------------------------------------------
// GEMM m=0:  Y0 = X0 (E x 640) * W0_g^T (640x640) + b0_g, scatter to rows l*l+l
// one wave -> 16 rows x 32 cols (two 16x16 N-tiles sharing the A fragment)
// ---------------------------------------------------------------------------
__global__ void gemm_m0(const _Float16* __restrict__ X0,
                        const _Float16* __restrict__ W,
                        const float* __restrict__ bias,
                        float* __restrict__ Ybuf) {
    const int K = (L_MAX + 1) * C_IN;    // 640
    const int N = (L_MAX + 1) * C_OUT;   // 640
    const int nPairs = N / 32;           // 20
    int wave = (blockIdx.x * blockDim.x + threadIdx.x) >> 5;
    int mt = wave / nPairs;
    int np = wave - mt * nPairs;
    if (mt >= N_EDGES / 16) return;
    int lane = threadIdx.x & 31, half = lane >> 4, l15 = lane & 15;
    int g = (mt * 16) / EDGES_PER_G;
    const _Float16* Wg = W + (size_t)g * N * K;
    int n0 = np * 32;
    const _Float16* Abase = X0 + (size_t)(mt * 16 + l15) * K + half * 8;
    const _Float16* B0base = Wg + (size_t)(n0 + l15) * K + half * 8;
    const _Float16* B1base = B0base + (size_t)16 * K;
    f32x8 c0 = {}, c1 = {};
    for (int k0 = 0; k0 < K; k0 += 32) {
        half16 a = loadA(Abase + k0);
        half16 b0 = loadB(B0base + k0);
        half16 b1 = loadB(B1base + k0);
        c0 = __builtin_amdgcn_wmma_f32_16x16x32_f16(false, a, false, b0, (short)0, c0, false, false);
        c1 = __builtin_amdgcn_wmma_f32_16x16x32_f16(false, a, false, b1, (short)0, c1, false, false);
    }
    int col0 = n0 + l15, col1 = col0 + 16;
    float bb0 = bias[g * N + col0], bb1 = bias[g * N + col1];
    int lc0 = col0 >> 7, cc0 = col0 & 127;
    int lc1 = col1 >> 7, cc1 = col1 & 127;
    int row0 = mt * 16 + half * 8;
#pragma unroll
    for (int r = 0; r < 8; ++r) {
        size_t e = (size_t)(row0 + r);
        Ybuf[(e * D_TOT + (lc0 * lc0 + lc0)) * C_OUT + cc0] = c0[r] + bb0;
        Ybuf[(e * D_TOT + (lc1 * lc1 + lc1)) * C_OUT + cc1] = c1[r] + bb1;
    }
}

// ---------------------------------------------------------------------------
// GEMM m>=1 with fused +/-m recombination:
//   per wave: one 16-row x 16-col (col in [0,H)) tile;
//   4 accumulators: {A_t0, A_t1} x {W_lo rows [0,H), W_hi rows [H,2H)}
//   yp = c00 - c11 -> row l*l+l-m ;  yn = c10 + c01 -> row l*l+l+m
// ---------------------------------------------------------------------------
__global__ void gemm_m(const _Float16* __restrict__ Xm,
                       const _Float16* __restrict__ W,
                       float* __restrict__ Ybuf, int m) {
    const int numl = L_MAX + 1 - m;
    const int H = numl * C_IN;           // = fi ; fo = 2*H
    const int K = H;
    int nTiles = H / 16;
    int wave = (blockIdx.x * blockDim.x + threadIdx.x) >> 5;
    int mt = wave / nTiles;
    int nt = wave - mt * nTiles;
    if (mt >= N_EDGES / 16) return;
    int lane = threadIdx.x & 31, half = lane >> 4, l15 = lane & 15;
    int g = (mt * 16) / EDGES_PER_G;
    const _Float16* Wg = W + (size_t)g * (2 * H) * K;
    int n0 = nt * 16;
    const _Float16* A0 = Xm + ((size_t)(mt * 16 + l15) * 2 + 0) * K + half * 8;
    const _Float16* A1 = Xm + ((size_t)(mt * 16 + l15) * 2 + 1) * K + half * 8;
    const _Float16* BL = Wg + (size_t)(n0 + l15) * K + half * 8;
    const _Float16* BH = Wg + (size_t)(H + n0 + l15) * K + half * 8;
    f32x8 c00 = {}, c01 = {}, c10 = {}, c11 = {};
    for (int k0 = 0; k0 < K; k0 += 32) {
        half16 a0 = loadA(A0 + k0);
        half16 a1 = loadA(A1 + k0);
        half16 bl = loadB(BL + k0);
        half16 bh = loadB(BH + k0);
        c00 = __builtin_amdgcn_wmma_f32_16x16x32_f16(false, a0, false, bl, (short)0, c00, false, false);
        c01 = __builtin_amdgcn_wmma_f32_16x16x32_f16(false, a0, false, bh, (short)0, c01, false, false);
        c10 = __builtin_amdgcn_wmma_f32_16x16x32_f16(false, a1, false, bl, (short)0, c10, false, false);
        c11 = __builtin_amdgcn_wmma_f32_16x16x32_f16(false, a1, false, bh, (short)0, c11, false, false);
    }
    int col = n0 + l15;
    int lj = col >> 7, c = col & 127;
    int l = m + lj;
    int row_p = l * l + l - m;
    int row_n = l * l + l + m;
    int row0 = mt * 16 + half * 8;
#pragma unroll
    for (int r = 0; r < 8; ++r) {
        size_t e = (size_t)(row0 + r);
        Ybuf[(e * D_TOT + row_p) * C_OUT + c] = c00[r] - c11[r];
        Ybuf[(e * D_TOT + row_n) * C_OUT + c] = c10[r] + c01[r];
    }
}

// ---------------------------------------------------------------------------
// Kernel: rotate back with D_l^T, write final fp32 output
// ---------------------------------------------------------------------------
__global__ void rot_back_kernel(const float* __restrict__ Ybuf,
                                const float* __restrict__ Dw,
                                float* __restrict__ out) {
    __shared__ float sD[2 * DW_STRIDE];
    int e0 = blockIdx.x * 2;
    for (int i = threadIdx.x; i < 2 * DW_STRIDE; i += blockDim.x)
        sD[i] = Dw[(size_t)e0 * DW_STRIDE + i];
    __syncthreads();
    int le = threadIdx.x >> 7;
    int c = threadIdx.x & 127;
    int e = e0 + le;
    const float* De = sD + le * DW_STRIDE;
    int dOff = 0;
    for (int l = 0; l <= L_MAX; ++l) {
        int d = 2 * l + 1;
        float yv[9];
        for (int j = 0; j < d; ++j)
            yv[j] = Ybuf[((size_t)e * D_TOT + l * l + j) * C_OUT + c];
        for (int i = 0; i < d; ++i) {
            float s = 0.f;
            for (int j = 0; j < d; ++j) s += De[dOff + j * d + i] * yv[j];  // D^T
            out[((size_t)e * D_TOT + l * l + i) * C_OUT + c] = s;
        }
        dOff += d * d;
    }
}

// ---------------------------------------------------------------------------
extern "C" void kernel_launch(void* const* d_in, const int* in_sizes, int n_in,
                              void* d_out, int out_size, void* d_ws, size_t ws_size,
                              hipStream_t stream) {
    (void)in_sizes; (void)out_size; (void)ws_size;
    const float* x     = (const float*)d_in[0];
    const float* alpha = (const float*)d_in[1];
    const float* beta  = (const float*)d_in[2];
    const float* gamma = (const float*)d_in[3];
    const float* coeff = (const float*)d_in[4];
    const float* w0_e  = (const float*)d_in[5];
    const float* b0_e  = (const float*)d_in[6];
    const float* w0_s  = (const float*)d_in[7];
    const float* b0_s  = (const float*)d_in[8];
    const float* wm_e[4];
    const float* wm_s[4];
    if (n_in >= 17) {
        for (int mi = 0; mi < 4; ++mi) {
            wm_e[mi] = (const float*)d_in[9 + mi];
            wm_s[mi] = (const float*)d_in[13 + mi];
        }
    } else {
        // lists passed concatenated
        const float* pe = (const float*)d_in[9];
        const float* ps = (const float*)d_in[10];
        size_t oe = 0, os = 0;
        for (int mi = 0; mi < 4; ++mi) {
            int numl = L_MAX - mi;                 // 4,3,2,1 for m=1..4
            size_t fi = (size_t)numl * C_IN, fo = (size_t)2 * numl * C_OUT;
            wm_e[mi] = pe + oe; wm_s[mi] = ps + os;
            oe += (size_t)N_EXPERTS * fo * fi;
            os += fo * fi;
        }
    }

    // workspace carve (256B aligned slabs)
    char* p = (char*)d_ws;
    auto alloc = [&](size_t bytes) -> void* {
        void* r = (void*)p;
        p += (bytes + 255) & ~(size_t)255;
        return r;
    };
    float*    gh   = (float*)alloc(GH_TOTAL * sizeof(float));
    float*    Dw   = (float*)alloc((size_t)N_EDGES * DW_STRIDE * sizeof(float));
    _Float16* X0   = (_Float16*)alloc((size_t)N_EDGES * (L_MAX + 1) * C_IN * sizeof(_Float16));
    _Float16* Xm[5] = {nullptr, nullptr, nullptr, nullptr, nullptr};
    for (int m = 1; m <= L_MAX; ++m) {
        size_t H = (size_t)(L_MAX + 1 - m) * C_IN;
        Xm[m] = (_Float16*)alloc((size_t)N_EDGES * 2 * H * sizeof(_Float16));
    }
    float*    Ybuf = (float*)alloc((size_t)N_EDGES * D_TOT * C_OUT * sizeof(float));
    _Float16* W0c  = (_Float16*)alloc((size_t)N_GRAPHS * 640 * 640 * sizeof(_Float16));
    float*    b0c  = (float*)alloc((size_t)N_GRAPHS * 640 * sizeof(float));
    _Float16* Wmc[5] = {nullptr, nullptr, nullptr, nullptr, nullptr};
    for (int m = 1; m <= L_MAX; ++m) {
        size_t H = (size_t)(L_MAX + 1 - m) * C_IN;
        Wmc[m] = (_Float16*)alloc((size_t)N_GRAPHS * 2 * H * H * sizeof(_Float16));
    }

    // 1) constants + Wigner matrices
    hipLaunchKernelGGL(init_ry_consts, dim3(1), dim3(32), 0, stream, gh);
    hipLaunchKernelGGL(wigner_kernel, dim3(N_EDGES / 256), dim3(256), 0, stream,
                       alpha, beta, gamma, gh, Dw);

    // 2) rotate + gather into f16 panels
    hipLaunchKernelGGL(rot_gather_kernel, dim3(N_EDGES / 2), dim3(256), 0, stream,
                       x, Dw, X0, Xm[1], Xm[2], Xm[3], Xm[4]);

    // 3) per-graph expert mixing
    {
        int sz = 640 * 640;
        hipLaunchKernelGGL(combine_w_f16, dim3((N_GRAPHS * sz + 255) / 256), dim3(256), 0, stream,
                           coeff, w0_e, w0_s, W0c, sz);
        hipLaunchKernelGGL(combine_b_f32, dim3((N_GRAPHS * 640 + 255) / 256), dim3(256), 0, stream,
                           coeff, b0_e, b0_s, b0c, 640);
        for (int m = 1; m <= L_MAX; ++m) {
            int H = (L_MAX + 1 - m) * C_IN;
            int szm = 2 * H * H;
            hipLaunchKernelGGL(combine_w_f16, dim3((N_GRAPHS * szm + 255) / 256), dim3(256), 0, stream,
                               coeff, wm_e[m - 1], wm_s[m - 1], Wmc[m], szm);
        }
    }

    // 4) WMMA GEMMs (blockDim 256 = 8 waves; grids are exact)
    {
        int waves0 = (N_EDGES / 16) * (640 / 32);       // 20480
        hipLaunchKernelGGL(gemm_m0, dim3(waves0 / 8), dim3(256), 0, stream, X0, W0c, b0c, Ybuf);
        for (int m = 1; m <= L_MAX; ++m) {
            int H = (L_MAX + 1 - m) * C_IN;
            int waves = (N_EDGES / 16) * (H / 16);
            hipLaunchKernelGGL(gemm_m, dim3(waves / 8), dim3(256), 0, stream,
                               Xm[m], Wmc[m], Ybuf, m);
        }
    }

    // 5) rotate back -> d_out
    hipLaunchKernelGGL(rot_back_kernel, dim3(N_EDGES / 2), dim3(256), 0, stream,
                       Ybuf, Dw, (float*)d_out);
}
